// CausalSelfAttention_51762945851891
// MI455X (gfx1250) — matrix operand order repaired and checked by reading, not compiled, not run on hardware
//
#include <hip/hip_runtime.h>
#include <hip/hip_bf16.h>

// ---------------- problem constants ----------------
#define B_ 2
#define T_ 2048
#define C_ 1024
#define H_ 16
#define D_ 64
#define C3_ (3 * C_)

typedef __bf16 bf16_t;
typedef __attribute__((ext_vector_type(16))) __bf16 v16bf;
typedef __attribute__((ext_vector_type(8)))  float  v8f;

static __device__ __forceinline__ bf16_t f2bf(float f) { return (bf16_t)f; }

static __device__ __forceinline__ v8f wmma_bf16(v16bf a, v16bf b, v8f c) {
  // D = A(16x32 bf16) x B(32x16 bf16) + C(16x16 f32)
  return __builtin_amdgcn_wmma_f32_16x16x32_bf16(
      false, a, false, b, (short)0, c, false, false);
}

// A-fragment (16x32, row-major source, row stride ld elements):
// lane m = lane&15, half = lane>>4
// element i : K = k0 + 16*half + i
static __device__ __forceinline__ v16bf load_afrag(const bf16_t* p, int ld,
                                                   int row0, int k0, int lane) {
  const int m = lane & 15, half = lane >> 4;
  const bf16_t* base = p + (size_t)(row0 + m) * ld + k0 + 16 * half;
  v16bf a;
#pragma unroll
  for (int i = 0; i < 16; ++i) { a[i] = base[i]; }
  return a;
}

// B-fragment (32x16): source holds B transposed, i.e. src[n][k] contiguous in k.
// lane n = n0 + (lane&15); element j : K = k0 + 16*half + j  (16 contiguous bf16)
static __device__ __forceinline__ v16bf load_bfrag_T(const bf16_t* pT, int ld,
                                                     int n0, int k0, int lane) {
  const int n = n0 + (lane & 15), half = lane >> 4;
  const bf16_t* base = pT + (size_t)n * ld + k0 + 16 * half;
  v16bf b;
#pragma unroll
  for (int j = 0; j < 16; ++j) b[j] = base[j];
  return b;
}

// ---------------- conversion kernels ----------------
__global__ void cvt_f32_to_bf16(const float* __restrict__ in,
                                bf16_t* __restrict__ out, int n) {
  int i = blockIdx.x * blockDim.x + threadIdx.x;
  int stride = gridDim.x * blockDim.x;
  for (; i < n; i += stride) out[i] = f2bf(in[i]);
}

// in: K x N (row-major fp32) -> out: N x K (row-major bf16)  (transpose+convert)
__global__ void cvtT_f32_to_bf16(const float* __restrict__ in,
                                 bf16_t* __restrict__ out, int K, int N) {
  long total = (long)K * N;
  long stride = (long)gridDim.x * blockDim.x;
  for (long i = blockIdx.x * (long)blockDim.x + threadIdx.x; i < total; i += stride) {
    long n = i / K, k = i % K;
    out[i] = f2bf(in[k * (long)N + n]);
  }
}

// ---------------- QKV GEMM + RoPE ----------------
// x_bf: (B*T) x C ; wqkvT: (3C) x C (transposed).  One wave -> 32 rows x 64 cols.
// col-group cg: [0,16)=q head cg, [16,32)=k head cg-16, [32,48)=v head cg-32.
__global__ void qkv_rope_kernel(const bf16_t* __restrict__ xb,
                                const bf16_t* __restrict__ wqkvT,
                                bf16_t* __restrict__ qout,   // (B,H,T,D)
                                bf16_t* __restrict__ kout,   // (B,H,T,D)
                                bf16_t* __restrict__ vTout)  // (B,H,D,T)
{
  const int lane = threadIdx.x & 31;
  const int w = blockIdx.x * (blockDim.x >> 5) + (threadIdx.x >> 5);
  const int rowtile = w / (C3_ / 64);
  const int cg = w % (C3_ / 64);
  if (rowtile >= (B_ * T_) / 32) return;
  const int row0 = rowtile * 32;
  const int n0 = cg * 64;

  v8f acc[2][4];
#pragma unroll
  for (int rt = 0; rt < 2; ++rt)
#pragma unroll
    for (int f = 0; f < 4; ++f)
#pragma unroll
      for (int r = 0; r < 8; ++r) acc[rt][f][r] = 0.f;

  for (int k0 = 0; k0 < C_; k0 += 32) {
    if (k0 + 32 < C_) {  // speculative prefetch of next weight tile -> global_prefetch_b8
      __builtin_prefetch(wqkvT + (size_t)(n0 + 2 * lane) * C_ + k0 + 32, 0, 3);
      __builtin_prefetch(wqkvT + (size_t)(n0 + 2 * lane + 1) * C_ + k0 + 32, 0, 3);
    }
    v16bf a0 = load_afrag(xb, C_, row0, k0, lane);
    v16bf a1 = load_afrag(xb, C_, row0 + 16, k0, lane);
#pragma unroll
    for (int f = 0; f < 4; ++f) {
      v16bf bb = load_bfrag_T(wqkvT, C_, n0 + 16 * f, k0, lane);
      acc[0][f] = wmma_bf16(a0, bb, acc[0][f]);
      acc[1][f] = wmma_bf16(a1, bb, acc[1][f]);
    }
  }

  const int n_in = lane & 15, half = lane >> 4;
  const int bidx = row0 >> 11;          // / T_
  const int tbase0 = row0 & (T_ - 1);

  if (cg < 32) {                        // q or k -> RoPE
    const bool isq = cg < 16;
    const int h = isq ? cg : cg - 16;
    bf16_t* outp = isq ? qout : kout;
    const float qscale = isq ? 0.125f : 1.0f;   // D^-0.5 folded into q
#pragma unroll
    for (int rt = 0; rt < 2; ++rt) {
      const int tbase = tbase0 + rt * 16;
#pragma unroll
      for (int f = 0; f < 4; ++f) {
        const int d = 16 * f + n_in;
        const float theta = __expf(-(float)(d >> 1) * 0.28782313662425574f); // ln(1e4)/32
#pragma unroll
        for (int r = 0; r < 8; ++r) {
          const int M = r + 8 * half;
          const float t = (float)(tbase + M);
          float cs, sn;
          __sincosf(t * theta, &sn, &cs);
          const float val = acc[rt][f][r];
          const float rot = (f < 2) ? -acc[rt][f + 2][r] : acc[rt][f - 2][r];
          const float o = (val * cs + rot * sn) * qscale;
          outp[(((size_t)bidx * H_ + h) * T_ + (tbase + M)) * D_ + d] = f2bf(o);
        }
      }
    }
  } else {                              // v -> store transposed (B,H,D,T)
    const int h = cg - 32;
#pragma unroll
    for (int rt = 0; rt < 2; ++rt) {
      const int tbase = tbase0 + rt * 16;
#pragma unroll
      for (int f = 0; f < 4; ++f) {
        const int d = 16 * f + n_in;
#pragma unroll
        for (int r = 0; r < 8; ++r) {
          const int M = r + 8 * half;
          vTout[(((size_t)bidx * H_ + h) * D_ + d) * T_ + (tbase + M)] =
              f2bf(acc[rt][f][r]);
        }
      }
    }
  }
}

// ---------------- flash attention ----------------
// one 32-thread block (single wave) per (b,h, 32-query tile):
// K/V fragments loaded once per 32-key chunk, reused by both 16-query subtiles.
__global__ void flash_kernel(const bf16_t* __restrict__ qb,  // (B,H,T,D), pre-scaled
                             const bf16_t* __restrict__ kb,  // (B,H,T,D)
                             const bf16_t* __restrict__ vT,  // (B,H,D,T)
                             bf16_t* __restrict__ yb)        // (B,T,C)
{
  __shared__ bf16_t Pst[32 * 32];

  const int lane = threadIdx.x & 31;
  const int n_in = lane & 15, half = lane >> 4;
  const int qi = blockIdx.x & 63;          // T/32 = 64 query tiles
  const int bh = blockIdx.x >> 6;
  const int b = bh >> 4, h = bh & 15;
  const int t0 = qi * 32;

  const bf16_t* qptr = qb + (size_t)bh * T_ * D_;
  const bf16_t* kptr = kb + (size_t)bh * T_ * D_;
  const bf16_t* vptr = vT + (size_t)bh * D_ * T_;

  v16bf qa[2][2];
#pragma unroll
  for (int qt = 0; qt < 2; ++qt) {
    qa[qt][0] = load_afrag(qptr, D_, t0 + qt * 16, 0, lane);
    qa[qt][1] = load_afrag(qptr, D_, t0 + qt * 16, 32, lane);
  }

  v8f acc[2][4];
  float rowm[2][8], rowl[2][8];
#pragma unroll
  for (int qt = 0; qt < 2; ++qt) {
#pragma unroll
    for (int f = 0; f < 4; ++f)
#pragma unroll
      for (int r = 0; r < 8; ++r) acc[qt][f][r] = 0.f;
#pragma unroll
    for (int r = 0; r < 8; ++r) { rowm[qt][r] = -1e30f; rowl[qt][r] = 0.f; }
  }

  const int nchunks = qi + 1;               // keys [0, t0+32) in 32-key chunks
  for (int c = 0; c < nchunks; ++c) {
    const int kbase = c * 32;

    // ---- K fragments for this chunk (shared by both query subtiles) ----
    v16bf kf[2][2];
#pragma unroll
    for (int tt = 0; tt < 2; ++tt) {
      const bf16_t* kr = kptr + (size_t)(kbase + tt * 16 + n_in) * D_ + 16 * half;
      kf[tt][0] = *(const v16bf*)(kr);        // d in [0,32)
      kf[tt][1] = *(const v16bf*)(kr + 32);   // d in [32,64)
    }

    // ---- per query subtile: S, causal mask, online softmax, P -> LDS ----
#pragma unroll
    for (int qt = 0; qt < 2; ++qt) {
      v8f s[2];
#pragma unroll
      for (int tt = 0; tt < 2; ++tt) {
        v8f z;
#pragma unroll
        for (int r = 0; r < 8; ++r) z[r] = 0.f;
        z = wmma_bf16(qa[qt][0], kf[tt][0], z);
        s[tt] = wmma_bf16(qa[qt][1], kf[tt][1], z);
      }
#pragma unroll
      for (int tt = 0; tt < 2; ++tt) {
        const int key = kbase + tt * 16 + n_in;
#pragma unroll
        for (int r = 0; r < 8; ++r) {
          const int query = t0 + qt * 16 + r + 8 * half;
          if (key > query) s[tt][r] = -1e30f;
        }
      }
      float esc[8];
#pragma unroll
      for (int r = 0; r < 8; ++r) {
        float lm = fmaxf(s[0][r], s[1][r]);
#pragma unroll
        for (int m = 1; m < 16; m <<= 1) lm = fmaxf(lm, __shfl_xor(lm, m, 16));
        const float mnew = fmaxf(rowm[qt][r], lm);
        esc[r] = __expf(rowm[qt][r] - mnew);
        rowm[qt][r] = mnew;
        const float p0 = __expf(s[0][r] - mnew);
        const float p1 = __expf(s[1][r] - mnew);
        float ls = p0 + p1;
#pragma unroll
        for (int m = 1; m < 16; m <<= 1) ls += __shfl_xor(ls, m, 16);
        rowl[qt][r] = rowl[qt][r] * esc[r] + ls;
        const int M = qt * 16 + r + 8 * half;
        Pst[M * 32 + n_in]      = f2bf(p0);
        Pst[M * 32 + 16 + n_in] = f2bf(p1);
      }
#pragma unroll
      for (int f = 0; f < 4; ++f)
#pragma unroll
        for (int r = 0; r < 8; ++r) acc[qt][f][r] *= esc[r];
    }
    __syncthreads();

    // ---- reload P as A-fragments ----
    v16bf pa[2];
#pragma unroll
    for (int qt = 0; qt < 2; ++qt) {
      const int base = (qt * 16 + n_in) * 32 + 16 * half;   // m = lane&15
#pragma unroll
      for (int i = 0; i < 16; ++i) {
        pa[qt][i] = Pst[base + i];
      }
    }

    // ---- acc += P . V  (V fragments shared by both query subtiles) ----
#pragma unroll
    for (int f = 0; f < 4; ++f) {
      const bf16_t* vr = vptr + (size_t)(16 * f + n_in) * T_ + kbase + 16 * half;
      v16bf vb = *(const v16bf*)vr;
      acc[0][f] = wmma_bf16(pa[0], vb, acc[0][f]);
      acc[1][f] = wmma_bf16(pa[1], vb, acc[1][f]);
    }
    __syncthreads();
  }

  // ---- normalize + store y (B,T,C) as bf16 ----
#pragma unroll
  for (int qt = 0; qt < 2; ++qt)
#pragma unroll
    for (int f = 0; f < 4; ++f) {
      const int d = 16 * f + n_in;
#pragma unroll
      for (int r = 0; r < 8; ++r) {
        const int M = qt * 16 + r + 8 * half;
        const float o = acc[qt][f][r] / rowl[qt][r];
        yb[((size_t)b * T_ + t0 + M) * C_ + h * D_ + d] = f2bf(o);
      }
    }
}

// ---------------- output projection GEMM (fp32 out) ----------------
__global__ void proj_kernel(const bf16_t* __restrict__ yb,     // (B*T) x C
                            const bf16_t* __restrict__ wprojT, // C x C (transposed)
                            float* __restrict__ out)           // (B*T) x C fp32
{
  const int lane = threadIdx.x & 31;
  const int w = blockIdx.x * (blockDim.x >> 5) + (threadIdx.x >> 5);
  const int rowtile = w / (C_ / 64);
  const int cg = w % (C_ / 64);
  if (rowtile >= (B_ * T_) / 32) return;
  const int row0 = rowtile * 32;
  const int n0 = cg * 64;

  v8f acc[2][4];
#pragma unroll
  for (int rt = 0; rt < 2; ++rt)
#pragma unroll
    for (int f = 0; f < 4; ++f)
#pragma unroll
      for (int r = 0; r < 8; ++r) acc[rt][f][r] = 0.f;

  for (int k0 = 0; k0 < C_; k0 += 32) {
    if (k0 + 32 < C_) {
      __builtin_prefetch(wprojT + (size_t)(n0 + 2 * lane) * C_ + k0 + 32, 0, 3);
      __builtin_prefetch(wprojT + (size_t)(n0 + 2 * lane + 1) * C_ + k0 + 32, 0, 3);
    }
    v16bf a0 = load_afrag(yb, C_, row0, k0, lane);
    v16bf a1 = load_afrag(yb, C_, row0 + 16, k0, lane);
#pragma unroll
    for (int f = 0; f < 4; ++f) {
      v16bf bb = load_bfrag_T(wprojT, C_, n0 + 16 * f, k0, lane);
      acc[0][f] = wmma_bf16(a0, bb, acc[0][f]);
      acc[1][f] = wmma_bf16(a1, bb, acc[1][f]);
    }
  }

  const int n_in = lane & 15, half = lane >> 4;
#pragma unroll
  for (int rt = 0; rt < 2; ++rt)
#pragma unroll
    for (int f = 0; f < 4; ++f)
#pragma unroll
      for (int r = 0; r < 8; ++r) {
        const int M = rt * 16 + r + 8 * half;
        out[(size_t)(row0 + M) * C_ + n0 + 16 * f + n_in] = acc[rt][f][r];
      }
}

// ---------------- launch ----------------
extern "C" void kernel_launch(void* const* d_in, const int* in_sizes, int n_in_,
                              void* d_out, int out_size, void* d_ws, size_t ws_size,
                              hipStream_t stream) {
  const float* x      = (const float*)d_in[0];   // (B,T,C)
  const float* w_qkv  = (const float*)d_in[1];   // (C,3C)
  const float* w_proj = (const float*)d_in[2];   // (C,C)
  float* out = (float*)d_out;

  char* ws = (char*)d_ws;
  size_t off = 0;
  bf16_t* xb     = (bf16_t*)(ws + off); off += (size_t)B_ * T_ * C_ * 2;   // 8 MB
  bf16_t* wqkvT  = (bf16_t*)(ws + off); off += (size_t)C3_ * C_ * 2;       // 6 MB
  bf16_t* wprojT = (bf16_t*)(ws + off); off += (size_t)C_ * C_ * 2;        // 2 MB
  bf16_t* qb     = (bf16_t*)(ws + off); off += (size_t)B_ * H_ * T_ * D_ * 2;
  bf16_t* kb     = (bf16_t*)(ws + off); off += (size_t)B_ * H_ * T_ * D_ * 2;
  bf16_t* vT     = (bf16_t*)(ws + off); off += (size_t)B_ * H_ * D_ * T_ * 2;
  bf16_t* yb     = (bf16_t*)(ws + off); off += (size_t)B_ * T_ * C_ * 2;

  // 1) precision conversion (+ weight transposes)
  cvt_f32_to_bf16<<<4096, 256, 0, stream>>>(x, xb, B_ * T_ * C_);
  cvtT_f32_to_bf16<<<4096, 256, 0, stream>>>(w_qkv, wqkvT, C_, C3_);
  cvtT_f32_to_bf16<<<2048, 256, 0, stream>>>(w_proj, wprojT, C_, C_);

  // 2) QKV GEMM + fused RoPE (+ q pre-scale, v transpose)
  {
    int waves = ((B_ * T_) / 32) * (C3_ / 64);   // 128 * 48 = 6144
    qkv_rope_kernel<<<waves / 4, 128, 0, stream>>>(xb, wqkvT, qb, kb, vT);
  }

  // 3) flash attention: one wave per (b,h, 32-query tile)
  flash_kernel<<<B_ * H_ * (T_ / 32), 32, 0, stream>>>(qb, kb, vT, yb);

  // 4) output projection -> fp32
  {
    int waves = ((B_ * T_) / 32) * (C_ / 64);    // 128 * 16 = 2048
    proj_kernel<<<waves / 4, 128, 0, stream>>>(yb, wprojT, out);
  }
}
